// DecoderLayer_55362128445827
// MI455X (gfx1250) — matrix operand order repaired and checked by reading, not compiled, not run on hardware
//
#include <hip/hip_runtime.h>

#define DM  1024
#define FF  2048
#define SEQ 2048
#define NB  2
#define NH  16
#define MR  (NB * SEQ)          // 4096 token rows

typedef unsigned short u16;
typedef __attribute__((ext_vector_type(16))) __bf16 v16bf;
typedef __attribute__((ext_vector_type(8)))  float  v8f;

union Frag { v16bf v; uint4 q[2]; u16 us[16]; };

__device__ inline u16 f2bf(float f) {
  unsigned int u = __float_as_uint(f);
  unsigned int r = u + 0x7FFFu + ((u >> 16) & 1u);   // round-to-nearest-even
  if ((u & 0x7F800000u) == 0x7F800000u) r = u;       // Inf/NaN: truncate
  return (u16)(r >> 16);
}

__device__ inline void lds_fence() { asm volatile("s_wait_dscnt 0" ::: "memory"); }

// ---------------------------------------------------------------- convert f32 -> bf16
__global__ __launch_bounds__(256) void cvt_bf16_kernel(const float* __restrict__ src,
                                                       u16* __restrict__ dst, int n) {
  int i = blockIdx.x * 256 + threadIdx.x;
  if (i < n) dst[i] = f2bf(src[i]);
}

// ---------------------------------------------------------------- LayerNorm -> bf16
__global__ __launch_bounds__(256) void ln_bf16_kernel(const float* __restrict__ x,
                                                      const float* __restrict__ g,
                                                      const float* __restrict__ b,
                                                      u16* __restrict__ out) {
  const int row = blockIdx.x, tid = threadIdx.x;
  const float* xr = x + (size_t)row * DM;
  float vals[4], s = 0.f, ss = 0.f;
#pragma unroll
  for (int i = 0; i < 4; ++i) {
    float v = xr[tid + i * 256];
    vals[i] = v; s += v; ss += v * v;
  }
  __shared__ float red[256], red2[256];
  red[tid] = s; red2[tid] = ss; __syncthreads();
  for (int st = 128; st > 0; st >>= 1) {
    if (tid < st) { red[tid] += red[tid + st]; red2[tid] += red2[tid + st]; }
    __syncthreads();
  }
  float mu   = red[0] * (1.0f / DM);
  float var  = red2[0] * (1.0f / DM) - mu * mu;
  float rstd = rsqrtf(var + 1e-5f);
  u16* orow = out + (size_t)row * DM;
#pragma unroll
  for (int i = 0; i < 4; ++i) {
    int c = tid + i * 256;
    orow[c] = f2bf((vals[i] - mu) * rstd * g[c] + b[c]);
  }
}

// ---------------------------------------------------------------- WMMA GEMM
// out[m,n] = dot_k(A[m,k], W[n,k]) + bias[n]  (x @ W.T + b), optional relu/residual.
// One wave computes a 32(M) x 64(N) tile: 2 M-subtiles x 4 N-subtiles, 8 accumulators.
// Ping-pong double buffering: loads for K-step i+1 are issued before the WMMAs of
// step i, so VMEM latency overlaps matrix math (partial s_wait_loadcnt instead of 0).
__global__ __launch_bounds__(256) void gemm_bf16_wmma(
    const u16* __restrict__ A, const u16* __restrict__ W,
    const float* __restrict__ bias, const float* __restrict__ resid,
    float* __restrict__ outF, u16* __restrict__ outB,
    int M, int N, int K, int relu) {
  const int lane = threadIdx.x & 31;
  const int hf   = lane >> 4;       // half-wave select
  const int l16  = lane & 15;
  const int wid  = blockIdx.x * 8 + (threadIdx.x >> 5);
  const int waves_n = N >> 6;
  const int wm = wid / waves_n;
  const int wn = wid - wm * waves_n;
  if (wm * 32 >= M) return;         // wave-uniform guard (grids are exact)

  const size_t arow0 = (size_t)(wm * 32 + l16) * K;
  const size_t arow1 = arow0 + (size_t)16 * K;
  const size_t wrow  = (size_t)(wn * 64 + l16) * K;   // + t*16*K per N-subtile

  v8f zf = {};
  v8f acc[2][4];
#pragma unroll
  for (int mb = 0; mb < 2; ++mb)
#pragma unroll
    for (int t = 0; t < 4; ++t) acc[mb][t] = zf;

  Frag a0[2], b0[4], a1[2], b1[4];

  auto loadA = [&](Frag* a, int kb) {
    a[0].q[0] = *(const uint4*)(A + arow0 + kb + 8 * hf);
    a[0].q[1] = *(const uint4*)(A + arow0 + kb + 16 + 8 * hf);
    a[1].q[0] = *(const uint4*)(A + arow1 + kb + 8 * hf);
    a[1].q[1] = *(const uint4*)(A + arow1 + kb + 16 + 8 * hf);
  };
  auto loadB = [&](Frag* b, int kb) {
#pragma unroll
    for (int t = 0; t < 4; ++t) {
      const uint4* pb = (const uint4*)(W + wrow + (size_t)t * 16 * K + kb + hf * 8);
      b[t].q[0] = pb[0]; b[t].q[1] = pb[2];
    }
  };
  auto mma8 = [&](Frag* a, Frag* b) {
#pragma unroll
    for (int t = 0; t < 4; ++t) {
      acc[0][t] = __builtin_amdgcn_wmma_f32_16x16x32_bf16(
          false, a[0].v, false, b[t].v, (short)0, acc[0][t], false, false);
      acc[1][t] = __builtin_amdgcn_wmma_f32_16x16x32_bf16(
          false, a[1].v, false, b[t].v, (short)0, acc[1][t], false, false);
    }
  };

  loadA(a0, 0); loadB(b0, 0);
  for (int kb = 0; kb < K; kb += 64) {                 // K is a multiple of 64
    // prefetch two steps ahead (speculative; stays inside the buffers)
    __builtin_prefetch(A + arow0 + kb + 128);
    __builtin_prefetch(W + wrow + kb + 128);
    loadA(a1, kb + 32); loadB(b1, kb + 32);            // next half-step in flight
    mma8(a0, b0);
    if (kb + 64 < K) { loadA(a0, kb + 64); loadB(b0, kb + 64); }
    mma8(a1, b1);
  }

#pragma unroll
  for (int mb = 0; mb < 2; ++mb) {
    const int rowb = wm * 32 + mb * 16 + 8 * hf;
#pragma unroll
    for (int t = 0; t < 4; ++t) {
      const int col = wn * 64 + t * 16 + l16;
      const float bc = bias ? bias[col] : 0.f;
#pragma unroll
      for (int r = 0; r < 8; ++r) {
        const int row = rowb + r;
        float v = acc[mb][t][r] + bc;
        if (relu) v = fmaxf(v, 0.f);
        if (resid) v += resid[(size_t)row * N + col];
        if (outF) outF[(size_t)row * N + col] = v;
        if (outB) outB[(size_t)row * N + col] = f2bf(v);
      }
    }
  }
}

// ---------------------------------------------------------------- Flash attention
// One wave per (batch, head, 16-query tile). Streams keys in blocks of 32 with
// online softmax; 8 WMMAs per key block. Cross-block software pipeline: the K
// fragments and raw V block for step j+1 are loaded while step j computes, so
// the S-WMMAs consume register-resident K fragments with no load stall.
// Per-wave LDS: V transposed (4KB) + P staging (1KB).
__global__ __launch_bounds__(256) void flash_attn_wmma(
    const u16* __restrict__ Qb, const u16* __restrict__ Kb,
    const u16* __restrict__ Vb, u16* __restrict__ Ob) {
  __shared__ u16 smem[8 * 2560];                       // 40 KB static LDS
  const int lane = threadIdx.x & 31;
  const int hf   = lane >> 4;
  const int l16  = lane & 15;
  const int wv   = threadIdx.x >> 5;
  u16* Vt = smem + wv * 2560;                          // Vt[d][key] : 64x32
  u16* Pl = Vt + 2048;                                 // P[m][key]  : 16x32

  const int wid = blockIdx.x * 8 + wv;                 // 0..4095
  const int qt  = wid & 127;
  const int h   = (wid >> 7) & 15;
  const int bi  = wid >> 11;
  const size_t rowbase = (size_t)bi * SEQ;

  // Preload Q fragments (16 queries x 64 dk = two 16x32 A fragments)
  const size_t qrow = (rowbase + qt * 16 + l16) * DM + h * 64;
  Frag qf[2];
#pragma unroll
  for (int f = 0; f < 2; ++f) {
    qf[f].q[0] = *(const uint4*)(Qb + qrow + f * 32 + 8 * hf);
    qf[f].q[1] = *(const uint4*)(Qb + qrow + f * 32 + 16 + 8 * hf);
  }

  v8f zf = {};
  v8f o[4];
#pragma unroll
  for (int t = 0; t < 4; ++t) o[t] = zf;
  float m_i[8], l_i[8];
#pragma unroll
  for (int r = 0; r < 8; ++r) { m_i[r] = -1e30f; l_i[r] = 0.f; }

  auto loadK = [&](Frag* kf, int j) {                  // 4 B-fragments of K block
#pragma unroll
    for (int t = 0; t < 2; ++t)
#pragma unroll
      for (int f = 0; f < 2; ++f) {
        const size_t krow = (rowbase + j * 32 + t * 16 + l16) * DM + h * 64;
        const uint4* pk = (const uint4*)(Kb + krow + f * 32 + hf * 8);
        kf[t * 2 + f].q[0] = pk[0]; kf[t * 2 + f].q[1] = pk[2];
      }
  };
  auto loadV = [&](uint4* vr, int j) {                 // raw V block: lane = key row
    const u16* vp = Vb + (rowbase + j * 32 + lane) * DM + h * 64;
#pragma unroll
    for (int c = 0; c < 8; ++c) vr[c] = ((const uint4*)vp)[c];
  };

  auto step = [&](Frag* kc, uint4* vc, Frag* kn, uint4* vn, int jn) {
    // issue next block's global loads first: they overlap all compute below
    loadK(kn, jn);
    loadV(vn, jn);

    // stage current V block (already in registers) transposed: Vt[d][key]
#pragma unroll
    for (int c = 0; c < 8; ++c) {
      union { uint4 q; u16 u[8]; } tt; tt.q = vc[c];
#pragma unroll
      for (int e = 0; e < 8; ++e) Vt[(c * 8 + e) * 32 + lane] = tt.u[e];
    }
    lds_fence();

    // S = (Q @ K^T) * 1/sqrt(dk) from register-resident K fragments
    v8f s[2];
#pragma unroll
    for (int t = 0; t < 2; ++t) {
      s[t] = zf;
#pragma unroll
      for (int f = 0; f < 2; ++f)
        s[t] = __builtin_amdgcn_wmma_f32_16x16x32_bf16(
            false, qf[f].v, false, kc[t * 2 + f].v, (short)0, s[t], false, false);
#pragma unroll
      for (int r = 0; r < 8; ++r) s[t][r] *= 0.125f;   // 1/sqrt(64)
    }

    // Online softmax per query row (row = r + 8*hf, cols across the 16-lane group)
#pragma unroll
    for (int r = 0; r < 8; ++r) {
      float mx = fmaxf(s[0][r], s[1][r]);
#pragma unroll
      for (int off = 1; off < 16; off <<= 1) mx = fmaxf(mx, __shfl_xor(mx, off, 32));
      const float mnew  = fmaxf(m_i[r], mx);
      const float alpha = __expf(m_i[r] - mnew);
      s[0][r] = __expf(s[0][r] - mnew);
      s[1][r] = __expf(s[1][r] - mnew);
      float rs = s[0][r] + s[1][r];
#pragma unroll
      for (int off = 1; off < 16; off <<= 1) rs += __shfl_xor(rs, off, 32);
      l_i[r] = l_i[r] * alpha + rs;
      m_i[r] = mnew;
      o[0][r] *= alpha; o[1][r] *= alpha; o[2][r] *= alpha; o[3][r] *= alpha;
    }

    // P (C-layout f32) -> LDS bf16 row-major [16 x 32]
#pragma unroll
    for (int r = 0; r < 8; ++r) {
      Pl[(r + 8 * hf) * 32 + l16]      = f2bf(s[0][r]);
      Pl[(r + 8 * hf) * 32 + 16 + l16] = f2bf(s[1][r]);
    }
    lds_fence();

    // A fragment of P (16x32), B fragments of V (32x16 per dk sub-tile)
    Frag pa;
    pa.q[0] = *(const uint4*)(Pl + l16 * 32 + 8 * hf);
    pa.q[1] = *(const uint4*)(Pl + l16 * 32 + 16 + 8 * hf);
#pragma unroll
    for (int t = 0; t < 4; ++t) {
      const uint4* pv = (const uint4*)(Vt + (t * 16 + l16) * 32 + hf * 8);
      Frag vb; vb.q[0] = pv[0]; vb.q[1] = pv[2];
      o[t] = __builtin_amdgcn_wmma_f32_16x16x32_bf16(
          false, pa.v, false, vb.v, (short)0, o[t], false, false);
    }
  };

  Frag  k0[4], k1[4];
  uint4 v0[8], v1[8];
  loadK(k0, 0); loadV(v0, 0);
  for (int j = 0; j < SEQ / 32; j += 2) {              // 64 blocks, ping-pong by 2
    step(k0, v0, k1, v1, j + 1);
    const int j2 = (j + 2 < SEQ / 32) ? j + 2 : j + 1; // clamp: stay in bounds
    step(k1, v1, k0, v0, j2);
  }

  // Normalize and write bf16 attention output [row, h*64 + col]
#pragma unroll
  for (int t = 0; t < 4; ++t)
#pragma unroll
    for (int r = 0; r < 8; ++r) {
      const size_t row = rowbase + qt * 16 + r + 8 * hf;
      const int col = h * 64 + t * 16 + l16;
      Ob[row * DM + col] = f2bf(o[t][r] / l_i[r]);
    }
}

// ---------------------------------------------------------------- launcher
extern "C" void kernel_launch(void* const* d_in, const int* in_sizes, int n_in,
                              void* d_out, int out_size, void* d_ws, size_t ws_size,
                              hipStream_t stream) {
  (void)in_sizes; (void)n_in; (void)out_size; (void)ws_size;
  const float* x     = (const float*)d_in[0];
  const float* ln1_g = (const float*)d_in[1];
  const float* ln1_b = (const float*)d_in[2];
  const float* wq    = (const float*)d_in[3];
  const float* bq    = (const float*)d_in[4];
  const float* wk    = (const float*)d_in[5];
  const float* bk    = (const float*)d_in[6];
  const float* wvw   = (const float*)d_in[7];
  const float* bv    = (const float*)d_in[8];
  const float* wo    = (const float*)d_in[9];
  const float* bo    = (const float*)d_in[10];
  const float* ln2_g = (const float*)d_in[11];
  const float* ln2_b = (const float*)d_in[12];
  const float* w1    = (const float*)d_in[13];
  const float* b1    = (const float*)d_in[14];
  const float* w2    = (const float*)d_in[15];
  const float* b2    = (const float*)d_in[16];
  float* out = (float*)d_out;

  char*  ws  = (char*)d_ws;
  size_t off = 0;
  auto alloc = [&](size_t bytes) -> void* {
    void* p = ws + off;
    off += (bytes + 255) & ~(size_t)255;
    return p;
  };
  u16* wq_b  = (u16*)alloc((size_t)DM * DM * 2);
  u16* wk_b  = (u16*)alloc((size_t)DM * DM * 2);
  u16* wv_b  = (u16*)alloc((size_t)DM * DM * 2);
  u16* wo_b  = (u16*)alloc((size_t)DM * DM * 2);
  u16* w1_b  = (u16*)alloc((size_t)FF * DM * 2);
  u16* w2_b  = (u16*)alloc((size_t)DM * FF * 2);
  u16* xn_b  = (u16*)alloc((size_t)MR * DM * 2);
  u16* q_b   = (u16*)alloc((size_t)MR * DM * 2);
  u16* k_b   = (u16*)alloc((size_t)MR * DM * 2);
  u16* v_b   = (u16*)alloc((size_t)MR * DM * 2);
  u16* at_b  = (u16*)alloc((size_t)MR * DM * 2);
  float* xmid = (float*)alloc((size_t)MR * DM * 4);
  u16* xn2_b = (u16*)alloc((size_t)MR * DM * 2);
  u16* h_b   = (u16*)alloc((size_t)MR * FF * 2);

  auto cvt = [&](const float* s, u16* d, int n) {
    cvt_bf16_kernel<<<(n + 255) / 256, 256, 0, stream>>>(s, d, n);
  };
  cvt(wq, wq_b, DM * DM);
  cvt(wk, wk_b, DM * DM);
  cvt(wvw, wv_b, DM * DM);
  cvt(wo, wo_b, DM * DM);
  cvt(w1, w1_b, FF * DM);
  cvt(w2, w2_b, DM * FF);

  auto gblocks = [](int M, int N) { return (M / 32) * (N / 64) / 8; };

  // x2 = LN1(x) (bf16)
  ln_bf16_kernel<<<MR, 256, 0, stream>>>(x, ln1_g, ln1_b, xn_b);
  // Q/K/V = x2 @ W.T + b (bf16)
  gemm_bf16_wmma<<<gblocks(MR, DM), 256, 0, stream>>>(xn_b, wq_b, bq, nullptr,
                                                      nullptr, q_b, MR, DM, DM, 0);
  gemm_bf16_wmma<<<gblocks(MR, DM), 256, 0, stream>>>(xn_b, wk_b, bk, nullptr,
                                                      nullptr, k_b, MR, DM, DM, 0);
  gemm_bf16_wmma<<<gblocks(MR, DM), 256, 0, stream>>>(xn_b, wv_b, bv, nullptr,
                                                      nullptr, v_b, MR, DM, DM, 0);
  // attention (flash, online softmax, cross-block pipelined)
  flash_attn_wmma<<<(NB * NH * (SEQ / 16)) / 8, 256, 0, stream>>>(q_b, k_b, v_b, at_b);
  // xmid = x + attn @ wo.T + bo (f32)
  gemm_bf16_wmma<<<gblocks(MR, DM), 256, 0, stream>>>(at_b, wo_b, bo, x,
                                                      xmid, nullptr, MR, DM, DM, 0);
  // x2 = LN2(xmid) (bf16)
  ln_bf16_kernel<<<MR, 256, 0, stream>>>(xmid, ln2_g, ln2_b, xn2_b);
  // h = relu(x2 @ w1.T + b1) (bf16)
  gemm_bf16_wmma<<<gblocks(MR, FF), 256, 0, stream>>>(xn2_b, w1_b, b1, nullptr,
                                                      nullptr, h_b, MR, FF, DM, 1);
  // out = xmid + h @ w2.T + b2 (f32)
  gemm_bf16_wmma<<<gblocks(MR, DM), 256, 0, stream>>>(h_b, w2_b, b2, xmid,
                                                      out, nullptr, MR, DM, FF, 0);
}
